// VQVAE_67645734912601
// MI455X (gfx1250) — matrix operand order repaired and checked by reading, not compiled, not run on hardware
//
#include <hip/hip_runtime.h>
#include <hip/hip_bf16.h>
#include <math.h>

// ---------------------------------------------------------------------------
// VQ-VAE forward on gfx1250 (MI455X).  wave32, WMMA f32_16x16x32_f16,
// double-buffered TDM tensor_load_to_lds / async global->LDS weight staging.
// ---------------------------------------------------------------------------

typedef __attribute__((ext_vector_type(16))) _Float16 v16h;
typedef __attribute__((ext_vector_type(2)))  _Float16 v2h;
typedef __attribute__((ext_vector_type(8)))  float    v8f;
typedef __attribute__((ext_vector_type(4)))  uint32_t v4u_;
typedef __attribute__((ext_vector_type(8)))  int      v8i_;
typedef __attribute__((ext_vector_type(4)))  int      v4i_;

#if defined(__HIP_DEVICE_COMPILE__) && __has_builtin(__builtin_amdgcn_tensor_load_to_lds)
#define HAVE_TDM 1
#else
#define HAVE_TDM 0
#endif

#define B_    64
#define HID_  128
#define KCB_  1024
#define H1_   31            // conv1 output spatial
#define H2_   30            // conv2 output spatial
#define M2_   (B_*H2_*H2_)  // 57600  (GEMM-2 / VQ rows)
#define M1_   (B_*H1_*H1_)  // 61504  (dec1 GEMM rows)
#define NZ1_  (B_*HID_*H1_*H1_)   // 7,872,512
#define NVQ_  (M2_*HID_)          // 7,372,800
#define NOUT_ (B_*3*64*64)        // 786,432

// ---------------------------------------------------------------------------
// encoder conv1: 3->128, k=4, s=2, valid.  Direct (K=48, too small for WMMA).
// ---------------------------------------------------------------------------
__global__ void k_conv1(const float* __restrict__ x, const float* __restrict__ w,
                        const float* __restrict__ b, float* __restrict__ z1) {
  int t = blockIdx.x * 256 + threadIdx.x;
  if (t >= NZ1_) return;
  int ox = t % H1_; int r = t / H1_;
  int oy = r % H1_; r /= H1_;
  int co = r % HID_; int bb = r / HID_;
  float s = b[co];
  const float* wr = w + co * 3 * 16;
#pragma unroll
  for (int ci = 0; ci < 3; ++ci) {
    const float* xp = x + ((bb * 3 + ci) * 64 + 2 * oy) * 64 + 2 * ox;
#pragma unroll
    for (int kh = 0; kh < 4; ++kh)
#pragma unroll
      for (int kw = 0; kw < 4; ++kw)
        s += xp[kh * 64 + kw] * wr[ci * 16 + kh * 4 + kw];
  }
  z1[t] = fmaxf(s, 0.f);
}

// ---------------------------------------------------------------------------
// weight packs -> f16 GEMM B matrices [K=512][N=128]
// ---------------------------------------------------------------------------
__global__ void k_packw(const float* __restrict__ W, _Float16* __restrict__ Bw,
                        int transposed) {
  int i = blockIdx.x * 256 + threadIdx.x;
  if (i >= 512 * 128) return;
  int n = i & 127, k = i >> 7;
  int ci = k >> 2, kh = (k >> 1) & 1, kw = k & 1;
  int src = transposed ? (((ci * 128 + n) * 2 + kh) * 2 + kw)
                       : (((n * 128 + ci) * 2 + kh) * 2 + kw);
  Bw[i] = (_Float16)W[src];
}

// ---------------------------------------------------------------------------
// codebook pack: f16 B fragments in the wave32 lane layout of
// v_wmma_f32_16x16x32_f16.  Bpk[((ntile*4 + kchunk)*32 + lane)*16 + e].
// ---------------------------------------------------------------------------
__global__ void k_packcb(const float* __restrict__ cb, _Float16* __restrict__ Bpk,
                         float* __restrict__ cnorm) {
  int i = blockIdx.x * 256 + threadIdx.x;
  if (i < KCB_) {
    float s = 0.f;
    const float* r = cb + i * HID_;
    for (int d = 0; d < HID_; ++d) s += r[d] * r[d];
    cnorm[i] = 0.5f * s;
  }
  if (i >= 64 * 4 * 32 * 16) return;
  int e = i & 15; int r2 = i >> 4;
  int lane = r2 & 31; r2 >>= 5;
  int kc = r2 & 3; int ntile = r2 >> 2;
  int hf = lane >> 4;
  int n = ntile * 16 + (lane & 15);
  int k = kc * 32 + ((e < 8) ? ((hf ? 8 : 0) + e) : ((hf ? 24 : 16) + (e - 8)));
  Bpk[i] = (_Float16)cb[n * HID_ + k];
}

// ---------------------------------------------------------------------------
// WMMA im2col GEMM + bias + relu, double-buffered LDS pipeline.
//  MODE 0: conv2   z1[b,128,31,31] -> z[b,128,30,30], tap (oy+kh, ox+kw)
//  MODE 1: convT1  q [b,128,30,30] -> y1[b,128,31,31], tap (oy-kh, ox-kw) pad0
// Block 256 thr / 8 waves.  Tile M=64, N=128; wave w owns cols [16w,16w+16)
// and 4 M-subtiles -> one B fragment feeds 4 WMMAs per 32-K chunk.
// Stage for chunk i+1 is issued before computing chunk i (TENSOR/ASYNC cnt
// overlap), so TDM/async latency hides behind the WMMAs.
// ---------------------------------------------------------------------------
template <int MODE>
__global__ void k_gemm(const float* __restrict__ In, const _Float16* __restrict__ Bw,
                       const float* __restrict__ bias, float* __restrict__ Out) {
  constexpr int HIN  = MODE ? H2_ : H1_;
  constexpr int HOUT = MODE ? H1_ : H2_;
  __shared__ _Float16 As[2][64][40];    // +8 halfs pad (64-bank LDS)
  __shared__ _Float16 Bs[2][32][136];   // 128 data halfs + 8 pad halfs per row
  const int tid = threadIdx.x;
  const int lane = tid & 31, wave = tid >> 5;
  const int m0 = blockIdx.x * 64;

  // per-thread A-staging slot geometry (loop-invariant decode hoisted)
  int aoff[4], lmA[4], lkpA[4], oyA[4];
  bool c0[4], c1[4];
#pragma unroll
  for (int s = 0; s < 4; ++s) {
    int i = tid + s * 256;            // 0..1023 half2 pairs
    int lm = i >> 4, lkp = i & 15;
    lmA[s] = lm; lkpA[s] = lkp;
    int m = m0 + lm;
    int ox = m % HOUT; int r = m / HOUT;
    int oy = r % HOUT; int bb = r / HOUT;
    oyA[s] = oy;
    if (MODE == 0) {
      aoff[s] = ((bb * HID_) * HIN + oy) * HIN + ox;
      c0[s] = c1[s] = true;
    } else {
      aoff[s] = ((bb * HID_) * HIN) * HIN + ox;   // + (ci*HIN+iy)*HIN at use
      c0[s] = (ox < HIN);      // kw=0 tap ix=ox
      c1[s] = (ox >= 1);       // kw=1 tap ix=ox-1
    }
  }

  auto stageB = [&](int buf, int kc) {
#if HAVE_TDM
    if (tid < 32) {   // one wave issues one TDM op (EXEC ignored by TDM)
      uint64_t ga = (uint64_t)(uintptr_t)(Bw + kc * 128);
      uint32_t la = (uint32_t)(uintptr_t)&Bs[buf][0][0];
      v4u_ g0 = { 1u,                                   // count=1, no gather
                  la,                                   // lds_addr
                  (uint32_t)ga,                         // global_addr[31:0]
                  (uint32_t)(ga >> 32) | (2u << 30) };  // addr[56:32] | type=2
      // data_size=2B(1), pad_enable, pad every 64 DW by 4 DW -> 136-half stride
      v8i_ g1 = { (int)((1u << 16) | (1u << 20) | (5u << 22) | (3u << 25)),
                  (int)(128u << 16),   // tensor_dim0 = 128
                  (int)(512u << 16),   // tensor_dim1 = 512
                  (int)(128u << 16),   // tile_dim0 = 128
                  32,                  // tile_dim1 = 32
                  128,                 // tensor_dim0_stride = 128
                  0, 0 };
      v4i_ gz = { 0, 0, 0, 0 };
#if __clang_major__ >= 23
      v8i_ gz8 = { 0, 0, 0, 0, 0, 0, 0, 0 };
      __builtin_amdgcn_tensor_load_to_lds(g0, g1, gz, gz, gz8, 0);
#else
      __builtin_amdgcn_tensor_load_to_lds(g0, g1, gz, gz, 0);
#endif
    }
#else
    int i0 = tid * 2;
#pragma unroll
    for (int j = 0; j < 2; ++j) {
      int i = i0 + j;                       // 0..511 b128 transfers
      int lk = i >> 4, ln8 = (i & 15) * 8;  // row, 8-half group
      uint64_t ga = (uint64_t)(uintptr_t)(Bw + (kc + lk) * 128 + ln8);
      uint32_t la = (uint32_t)(uintptr_t)&Bs[buf][lk][ln8];
      asm volatile("global_load_async_to_lds_b128 %0, %1, off"
                   :: "v"(la), "v"(ga) : "memory");
    }
#endif
  };

  auto stageA = [&](int buf, int kc) {
#pragma unroll
    for (int s = 0; s < 4; ++s) {
      int k = kc + 2 * lkpA[s];
      int ci = k >> 2, kh = (k >> 1) & 1;
      float v0, v1;
      if (MODE == 0) {
        const float* p = In + aoff[s] + (ci * HIN + kh) * HIN;
        v0 = p[0]; v1 = p[1];
      } else {
        int iy = oyA[s] - kh;
        bool yok = (iy >= 0) && (iy < HIN);
        int base = aoff[s] + (ci * HIN + iy) * HIN;
        v0 = (yok && c0[s]) ? In[base] : 0.f;
        v1 = (yok && c1[s]) ? In[base - 1] : 0.f;
      }
      v2h p2; p2.x = (_Float16)v0; p2.y = (_Float16)v1;
      *(v2h*)&As[buf][lmA[s]][2 * lkpA[s]] = p2;
    }
  };

#if HAVE_TDM
#define STAGE_WAIT() __builtin_amdgcn_s_wait_tensorcnt(0)
#else
#define STAGE_WAIT() asm volatile("s_wait_asynccnt 0x0" ::: "memory")
#endif

  v8f acc[4];
  v8f zero = {};
#pragma unroll
  for (int t = 0; t < 4; ++t) acc[t] = zero;

  // pipeline prologue
  stageB(0, 0);
  stageA(0, 0);
  STAGE_WAIT();
  __syncthreads();

  const int hf = lane >> 4, nl = lane & 15;
  for (int kc = 0; kc < 512; kc += 32) {
    const int cur = (kc >> 5) & 1, nxt = cur ^ 1;
    if (kc + 32 < 512) {
      stageB(nxt, kc + 32);                 // async/TDM into other buffer
      stageA(nxt, kc + 32);
      __builtin_prefetch(Bw + (kc + 64 < 512 ? (kc + 64) : kc) * 128, 0, 0);
    }
    // ---- compute on current buffer: 1 B fragment -> 4 WMMAs ----
    v16h bf;
#pragma unroll
    for (int e = 0; e < 16; ++e) bf[e] = Bs[cur][(e < 8) ? ((hf ? 8 : 0) + e) : ((hf ? 24 : 16) + (e - 8))][wave * 16 + nl];
#pragma unroll
    for (int t = 0; t < 4; ++t) {
      v16h a;
#pragma unroll
      for (int e = 0; e < 16; ++e) {
        int vg = e >> 1;
        int kb = (vg < 4) ? ((hf ? 8 : 0) + 2 * vg) : ((hf ? 24 : 16) + 2 * (vg - 4));
        a[e] = As[cur][16 * t + nl][kb + (e & 1)];
      }
      acc[t] = __builtin_amdgcn_wmma_f32_16x16x32_f16(false, a, false, bf,
                                                      (short)0, acc[t], false, false);
    }
    STAGE_WAIT();        // the stage issued this iteration (targets nxt)
    __syncthreads();
  }
#undef STAGE_WAIT

  // ---- epilogue: bias + relu, scatter to NCHW ----
  const int col = wave * 16 + nl;
  const float bcol = bias[col];
#pragma unroll
  for (int t = 0; t < 4; ++t) {
#pragma unroll
    for (int r = 0; r < 8; ++r) {
      int m = m0 + 16 * t + (hf ? 8 : 0) + r;
      float v = fmaxf(acc[t][r] + bcol, 0.f);
      int ox = m % HOUT; int rr = m / HOUT; int oy = rr % HOUT; int bb = rr / HOUT;
      Out[((bb * HID_ + col) * HOUT + oy) * HOUT + ox] = v;
    }
  }
}

// ---------------------------------------------------------------------------
// VQ: scores = flat @ cb^T - 0.5||c||^2, argmax per row == argmin distance.
// Block 256 thr / 8 waves, 16 rows, full N=1024; 32 WMMAs/wave.
// ---------------------------------------------------------------------------
__global__ void k_vq(const float* __restrict__ Z, const _Float16* __restrict__ Bpk,
                     const float* __restrict__ cnorm, int* __restrict__ Idx) {
  const int tid = threadIdx.x, lane = tid & 31, wave = tid >> 5;
  const int hf = lane >> 4, nl = lane & 15;
  const int m0 = blockIdx.x * 16;
  const v16h* Bp = (const v16h*)Bpk;
  const float* Arow = Z + (size_t)(m0 + nl) * HID_;

  v8f acc[8];
  v8f zero = {};
#pragma unroll
  for (int t = 0; t < 8; ++t) acc[t] = zero;

#pragma unroll
  for (int kc = 0; kc < 4; ++kc) {
    v16h a;
#pragma unroll
    for (int e = 0; e < 16; ++e) {
      int vg = e >> 1;
      int kb = (vg < 4) ? ((hf ? 8 : 0) + 2 * vg) : ((hf ? 24 : 16) + 2 * (vg - 4));
      a[e] = (_Float16)Arow[kc * 32 + kb + (e & 1)];
    }
#pragma unroll
    for (int nt = 0; nt < 8; ++nt) {
      int ntile = wave * 8 + nt;
      v16h bf = Bp[(ntile * 4 + kc) * 32 + lane];
      acc[nt] = __builtin_amdgcn_wmma_f32_16x16x32_f16(false, a, false, bf,
                                                       (short)0, acc[nt], false, false);
    }
  }

  float bestv[8]; int besti[8];
#pragma unroll
  for (int r = 0; r < 8; ++r) { bestv[r] = -3.4e38f; besti[r] = 0; }
#pragma unroll
  for (int nt = 0; nt < 8; ++nt) {
    int n = (wave * 8 + nt) * 16 + nl;
    float cn = cnorm[n];
#pragma unroll
    for (int r = 0; r < 8; ++r) {
      float s = acc[nt][r] - cn;
      if (s > bestv[r] || (s == bestv[r] && n < besti[r])) { bestv[r] = s; besti[r] = n; }
    }
  }
#pragma unroll
  for (int msk = 8; msk >= 1; msk >>= 1) {
#pragma unroll
    for (int r = 0; r < 8; ++r) {
      float ov = __shfl_xor(bestv[r], msk, 32);
      int   oi = __shfl_xor(besti[r], msk, 32);
      if (ov > bestv[r] || (ov == bestv[r] && oi < besti[r])) { bestv[r] = ov; besti[r] = oi; }
    }
  }
  __shared__ float cv[16][8];
  __shared__ int   cidx[16][8];
  if (nl == 0) {
#pragma unroll
    for (int r = 0; r < 8; ++r) {
      int row = hf * 8 + r;
      cv[row][wave] = bestv[r];
      cidx[row][wave] = besti[r];
    }
  }
  __syncthreads();
  if (tid < 16) {
    float bv = cv[tid][0]; int bi = cidx[tid][0];
#pragma unroll
    for (int w = 1; w < 8; ++w) {
      float v = cv[tid][w]; int i2 = cidx[tid][w];
      if (v > bv || (v == bv && i2 < bi)) { bv = v; bi = i2; }
    }
    Idx[m0 + tid] = bi;
  }
}

// ---------------------------------------------------------------------------
// quantize gather + loss accumulation.  loss = (1+BETA)*mean((q-z)^2)
// ---------------------------------------------------------------------------
__global__ void k_vqpost(const float* __restrict__ Z, const int* __restrict__ Idx,
                         const float* __restrict__ cb, float* __restrict__ Q,
                         float* __restrict__ lossSum) {
  __shared__ float red[256];
  int t = blockIdx.x * 256 + threadIdx.x;
  float d2 = 0.f;
  if (t < NVQ_) {
    int m = t >> 7, d = t & 127;
    float q = cb[Idx[m] * HID_ + d];
    float z = Z[t];
    Q[t] = q;
    float df = q - z;
    d2 = df * df;
  }
  red[threadIdx.x] = d2;
  __syncthreads();
  for (int s = 128; s > 0; s >>= 1) {
    if (threadIdx.x < s) red[threadIdx.x] += red[threadIdx.x + s];
    __syncthreads();
  }
  if (threadIdx.x == 0) atomicAdd(lossSum, red[0]);
}

__global__ void k_hist(const int* __restrict__ Idx, float* __restrict__ counts) {
  int m = blockIdx.x * 256 + threadIdx.x;
  if (m < M2_) atomicAdd(&counts[Idx[m]], 1.0f);
}

__global__ void k_init(float* __restrict__ counts, float* __restrict__ lossSum) {
  int t = blockIdx.x * 256 + threadIdx.x;
  if (t < KCB_) counts[t] = 0.f;
  if (t == 0) lossSum[0] = 0.f;
}

__global__ void k_final(const float* __restrict__ counts,
                        const float* __restrict__ lossSum, float* __restrict__ out) {
  __shared__ float red[1024];
  int t = threadIdx.x;
  float p = counts[t] * (1.0f / (float)M2_);
  red[t] = p * logf(p + 1e-10f);
  __syncthreads();
  for (int s = 512; s > 0; s >>= 1) {
    if (t < s) red[t] += red[t + s];
    __syncthreads();
  }
  if (t == 0) {
    out[NOUT_]     = 1.25f * lossSum[0] * (1.0f / (float)NVQ_);  // loss
    out[NOUT_ + 1] = expf(-red[0]);                              // perplexity
  }
}

// ---------------------------------------------------------------------------
// decoder convT2: 128->3, k=4, s=2, to [64,3,64,64].  Direct (N=3); the 24 KB
// weight tensor is staged to LDS once per block with async global->LDS copies.
// ---------------------------------------------------------------------------
__global__ void k_dec2(const float* __restrict__ Y, const float* __restrict__ w,
                       const float* __restrict__ b, float* __restrict__ out) {
  __shared__ float wl[HID_ * 3 * 16];   // 24,576 B
  for (int i = threadIdx.x; i < (HID_ * 3 * 16) / 4; i += 256) {
    uint64_t ga = (uint64_t)(uintptr_t)(w + i * 4);
    uint32_t la = (uint32_t)(uintptr_t)&wl[i * 4];
    asm volatile("global_load_async_to_lds_b128 %0, %1, off"
                 :: "v"(la), "v"(ga) : "memory");
  }
  asm volatile("s_wait_asynccnt 0x0" ::: "memory");
  __syncthreads();

  int t = blockIdx.x * 256 + threadIdx.x;
  if (t >= NOUT_) return;
  int ox = t & 63; int r = t >> 6;
  int oy = r & 63; r >>= 6;
  int c = r % 3; int bb = r / 3;
  float s = b[c];
  for (int kh = 0; kh < 4; ++kh) {
    int iy2 = oy - kh;
    if (iy2 < 0 || (iy2 & 1)) continue;
    int iy = iy2 >> 1;
    if (iy >= H1_) continue;
    for (int kw = 0; kw < 4; ++kw) {
      int ix2 = ox - kw;
      if (ix2 < 0 || (ix2 & 1)) continue;
      int ix = ix2 >> 1;
      if (ix >= H1_) continue;
      const float* yp = Y + ((size_t)bb * HID_ * H1_ + iy) * H1_ + ix;
      const float* wp = wl + (c * 4 + kh) * 4 + kw;
      float acc = 0.f;
      for (int ci = 0; ci < HID_; ++ci)
        acc += yp[ci * (H1_ * H1_)] * wp[ci * 48];
      s += acc;
    }
  }
  out[t] = s;
}

// ---------------------------------------------------------------------------
extern "C" void kernel_launch(void* const* d_in, const int* in_sizes, int n_in,
                              void* d_out, int out_size, void* d_ws, size_t ws_size,
                              hipStream_t stream) {
  const float* x        = (const float*)d_in[0];
  const float* enc_w1   = (const float*)d_in[1];
  const float* enc_b1   = (const float*)d_in[2];
  const float* enc_w2   = (const float*)d_in[3];
  const float* enc_b2   = (const float*)d_in[4];
  const float* codebook = (const float*)d_in[5];
  const float* dec_w1   = (const float*)d_in[6];
  const float* dec_b1   = (const float*)d_in[7];
  const float* dec_w2   = (const float*)d_in[8];
  const float* dec_b2   = (const float*)d_in[9];
  float* out = (float*)d_out;

  char* ws = (char*)d_ws;
  size_t off = 0;
  auto carve = [&](size_t bytes) {
    char* p = ws + off;
    off = (off + bytes + 255) & ~(size_t)255;
    return p;
  };
  float*    z1      = (float*)carve((size_t)NZ1_ * 4);  // conv1 out; reused as y1
  float*    z       = (float*)carve((size_t)NVQ_ * 4);  // conv2 out
  float*    q       = (float*)carve((size_t)NVQ_ * 4);  // quantized
  int*      idx     = (int*)carve((size_t)M2_ * 4);
  _Float16* Bw2     = (_Float16*)carve(512 * 128 * 2);
  _Float16* Bd1     = (_Float16*)carve(512 * 128 * 2);
  _Float16* Bpk     = (_Float16*)carve(64 * 4 * 32 * 16 * 2);
  float*    cnorm   = (float*)carve(KCB_ * 4);
  float*    counts  = (float*)carve(KCB_ * 4);
  float*    lossSum = (float*)carve(256);
  float*    y1      = z1;  // z1 dead after conv2

  k_init<<<4, 256, 0, stream>>>(counts, lossSum);
  k_packw<<<(512 * 128 + 255) / 256, 256, 0, stream>>>(enc_w2, Bw2, 0);
  k_packw<<<(512 * 128 + 255) / 256, 256, 0, stream>>>(dec_w1, Bd1, 1);
  k_packcb<<<(64 * 4 * 32 * 16 + 255) / 256, 256, 0, stream>>>(codebook, Bpk, cnorm);

  k_conv1<<<(NZ1_ + 255) / 256, 256, 0, stream>>>(x, enc_w1, enc_b1, z1);
  k_gemm<0><<<M2_ / 64, 256, 0, stream>>>(z1, Bw2, enc_b2, z);
  k_vq<<<M2_ / 16, 256, 0, stream>>>(z, Bpk, cnorm, idx);
  k_vqpost<<<(NVQ_ + 255) / 256, 256, 0, stream>>>(z, idx, codebook, q, lossSum);
  k_hist<<<(M2_ + 255) / 256, 256, 0, stream>>>(idx, counts);
  k_gemm<1><<<M1_ / 64, 256, 0, stream>>>(q, Bd1, dec_b1, y1);
  k_dec2<<<(NOUT_ + 255) / 256, 256, 0, stream>>>(y1, dec_w2, dec_b2, out);
  k_final<<<1, 1024, 0, stream>>>(counts, lossSum, out);
}
